// DenseMPS_77103252898443
// MI455X (gfx1250) — compile-verified, matrix operand order checked by reading, not run on hardware
//
#include <hip/hip_runtime.h>
#include <hip/hip_bf16.h>

typedef __attribute__((ext_vector_type(16))) _Float16 v16h;
typedef __attribute__((ext_vector_type(8)))  float    v8f;

#define BATCH 64
#define DBOND 32

// ---------------------------------------------------------------------------
// WMMA fragment helpers (layouts per CDNA5 ISA 7.12.2, wave32)
// A 16x32 f16: lane L -> row M = L&15 ; element e -> K = base + (e>>3)*16 + (e&7),
//   base = (L<16 ? 0 : 8)
// ---------------------------------------------------------------------------
__device__ __forceinline__ v16h lds_a_frag(const _Float16* sA, int ld, int mb,
                                           int kb, int lane) {
  v16h a;
  int m  = mb + (lane & 15);
  int k0 = kb + ((lane & 16) ? 8 : 0);
  const _Float16* row = sA + m * ld + k0;
#pragma unroll
  for (int e = 0; e < 8; ++e) {
    a[e]     = row[e];
    a[e + 8] = row[e + 16];
  }
  return a;
}

// B 32x16 f16 from global f32 matrix B[k][n] (row stride ldb), with K guard.
__device__ __forceinline__ v16h glb_b_frag(const float* B, int ldb, int kb,
                                           int nb, int K, int lane) {
  v16h r;
  int n  = nb + (lane & 15);
  int k0 = kb + ((lane & 16) ? 8 : 0);
#pragma unroll
  for (int e = 0; e < 8; ++e) {
    int ka = k0 + e;
    int kc = k0 + 16 + e;
    r[e]     = (_Float16)((ka < K) ? B[(size_t)ka * ldb + n] : 0.0f);
    r[e + 8] = (_Float16)((kc < K) ? B[(size_t)kc * ldb + n] : 0.0f);
  }
  return r;
}

// ---------------------------------------------------------------------------
// Gather kernels: build per-level site tensors (b, C, S=4, P) from feature maps
// ---------------------------------------------------------------------------
__global__ __launch_bounds__(256) void gather1_k(const float* __restrict__ x,
                                                 float* __restrict__ xin1) {
  int idx = blockIdx.x * 256 + threadIdx.x;          // (b,16,4,256) row-major
  if (idx >= BATCH * 16 * 4 * 256) return;
  int p  = idx & 255;
  int s  = (idx >> 8) & 3;
  int ch = (idx >> 10) & 15;
  int b  = idx >> 14;
  int u2 = p >> 4, v2 = p & 15;
  int c0 = ((ch & 3) << 2) | s;
  int i2 = (ch >> 3) & 1, j2 = (ch >> 2) & 1;
  int A  = i2 * 16 + u2;
  int Cc = j2 * 16 + v2;
  int L1 = A * 512 + Cc * 16 + c0;
  int v  = L1 & 31;
  int u  = (L1 >> 5) & 31;
  int ij = L1 >> 10;
  int i  = ij >> 2, j = ij & 3;
  xin1[idx] = x[(size_t)b * 16384 + (i * 32 + u) * 128 + (j * 32 + v)];
}

__global__ __launch_bounds__(256) void gather2_k(const float* __restrict__ cat1,
                                                 float* __restrict__ xin2) {
  int idx = blockIdx.x * 256 + threadIdx.x;          // (b,64,4,64)
  if (idx >= BATCH * 64 * 4 * 64) return;
  int p = idx & 63;
  int s = (idx >> 6) & 3;
  int c = (idx >> 8) & 63;
  int b = idx >> 14;
  int h = (s >> 1) * 8 + (p >> 3);
  int w = (s & 1) * 8 + (p & 7);
  xin2[idx] = cat1[((size_t)b * 64 + c) * 256 + h * 16 + w];
}

__global__ __launch_bounds__(256) void gather3_k(const float* __restrict__ cat2,
                                                 float* __restrict__ xin3) {
  int idx = blockIdx.x * 256 + threadIdx.x;          // (b,128,4,16)
  if (idx >= BATCH * 128 * 4 * 16) return;
  int p = idx & 15;
  int s = (idx >> 4) & 3;
  int c = (idx >> 6) & 127;
  int b = idx >> 13;
  int h = (s >> 1) * 4 + (p >> 2);
  int w = (s & 1) * 4 + (p & 3);
  xin3[idx] = cat2[((size_t)b * 128 + c) * 64 + h * 8 + w];
}

// ---------------------------------------------------------------------------
// Fused per-level MPS kernel: one block per patch p, all 64 batch rows.
// For site s: M_s = phi_s[b,F] x cores[p,s][F,1024] via WMMA f16 16x16x32,
// immediately folded into r_new[b,j] += r_old[b,i] * M_s[b,i,j] (ds atomics).
// Final projection r[b,:] x wr[p] -> yraw (b, O, P).
// ---------------------------------------------------------------------------
template <int CCH, int PP, int OO, int FPAD>
__global__ __launch_bounds__(512) void mps_level_k(
    const float* __restrict__ xin, const float* __restrict__ cores,
    const float* __restrict__ vl, const float* __restrict__ wr,
    float* __restrict__ yraw) {
  constexpr int F = 3 * CCH;
  __shared__ _Float16 sPhi[BATCH * FPAD];
  __shared__ float    sR[2][BATCH * DBOND];
  const int p = blockIdx.x;
  const int tid = threadIdx.x;
  const int lane = tid & 31;
  const int wave = tid >> 5;

  for (int s = 0; s < 4; ++s) {
    float* rNew = sR[s & 1];
    float* rOld = sR[(s + 1) & 1];
    // stage phi (zero-padded to FPAD) and clear accumulator
    for (int i = tid; i < BATCH * FPAD; i += 512) {
      int b = i / FPAD, f = i % FPAD;
      float val = 0.0f;
      if (f < F) {
        int c = f % CCH;
        float xv = xin[(((size_t)b * CCH + c) * 4 + s) * PP + p];
        val = (f < CCH) ? xv : ((f < 2 * CCH) ? (1.0f - xv) : xv * xv);
      }
      sPhi[i] = (_Float16)val;
    }
    for (int i = tid; i < BATCH * DBOND; i += 512) rNew[i] = 0.0f;
    __syncthreads();

    const float* Bs = cores + (((size_t)p * 4 + s) * F) * 1024;
    // 4 m-tiles x 64 n-tiles = 256 C tiles over 16 waves
    for (int t = wave; t < 256; t += 16) {
      int mt = t >> 6;
      int nt = t & 63;
      v8f c = {};
#pragma unroll
      for (int kb = 0; kb < FPAD; kb += 32) {
        v16h af = lds_a_frag(sPhi, FPAD, mt * 16, kb, lane);
        v16h bf = glb_b_frag(Bs, 1024, kb, nt * 16, F, lane);
        c = __builtin_amdgcn_wmma_f32_16x16x32_f16(false, af, false, bf,
                                                   (short)0, c, false, false);
      }
      // C tile covers fixed i = nt>>1, j-halfblock (nt&1)*16
      int i_ = nt >> 1;
      int j  = ((nt & 1) << 4) + (lane & 15);
      int hb = (lane & 16) ? 8 : 0;
#pragma unroll
      for (int v = 0; v < 8; ++v) {
        int b_ = mt * 16 + v + hb;
        float scale = (s == 0) ? vl[p * DBOND + i_] : rOld[b_ * DBOND + i_];
        atomicAdd(&rNew[b_ * DBOND + j], scale * c[v]);
      }
    }
    __syncthreads();
  }

  const float* rF = sR[1];  // s=3 wrote sR[1]
  for (int i = tid; i < BATCH * OO; i += 512) {
    int b = i / OO, o = i % OO;
    float acc = 0.0f;
#pragma unroll 8
    for (int j = 0; j < DBOND; ++j)
      acc += rF[b * DBOND + j] * wr[((size_t)p * DBOND + j) * OO + o];
    yraw[((size_t)b * OO + o) * PP + p] = acc;
  }
}

// ---------------------------------------------------------------------------
// BatchNorm of MPS output (b,O,P) over (b,p); write into concat buffer
// (b, Ctot, P) at channel offset coff.
// ---------------------------------------------------------------------------
__global__ __launch_bounds__(256) void bn_mps_to_cat_k(
    const float* __restrict__ y, float* __restrict__ cat, int O, int P,
    int Ctot, int coff) {
  int o = blockIdx.x, tid = threadIdx.x;
  int N = BATCH * P;
  float s = 0.f, s2 = 0.f;
  for (int i = tid; i < N; i += 256) {
    int b = i / P, p = i % P;
    float v = y[((size_t)b * O + o) * P + p];
    s += v; s2 += v * v;
  }
  __shared__ float red[256], red2[256];
  red[tid] = s; red2[tid] = s2;
  __syncthreads();
  for (int st = 128; st > 0; st >>= 1) {
    if (tid < st) { red[tid] += red[tid + st]; red2[tid] += red2[tid + st]; }
    __syncthreads();
  }
  float m = red[0] / N;
  float var = red2[0] / N - m * m;
  float inv = rsqrtf(var + 1e-5f);
  for (int i = tid; i < N; i += 256) {
    int b = i / P, p = i % P;
    float v = y[((size_t)b * O + o) * P + p];
    cat[((size_t)b * Ctot + coff + o) * P + p] = (v - m) * inv;
  }
}

// ---------------------------------------------------------------------------
// Uniform kxk average pooling: (b,C,H,W) -> (b,C,H/k,W/k)
// ---------------------------------------------------------------------------
__global__ __launch_bounds__(256) void pool_k(const float* __restrict__ in,
                                              float* __restrict__ out, int C,
                                              int H, int W, int k) {
  int OH = H / k, OW = W / k;
  int total = BATCH * C * OH * OW;
  int idx = blockIdx.x * 256 + threadIdx.x;
  if (idx >= total) return;
  int ow = idx % OW;
  int oh = (idx / OW) % OH;
  int c  = (idx / (OW * OH)) % C;
  int b  = idx / (OW * OH * C);
  const float* base = in + ((size_t)b * C + c) * H * W + (oh * k) * W + ow * k;
  float acc = 0.f;
  for (int u = 0; u < k; ++u)
    for (int v = 0; v < k; ++v) acc += base[u * W + v];
  out[idx] = acc / (float)(k * k);
}

// ---------------------------------------------------------------------------
// BN (per-channel over b,h,w) + ReLU + write into concat buffer with optional
// channel replication (rep>1 used for the broadcast z at level 1).
// ---------------------------------------------------------------------------
__global__ __launch_bounds__(256) void bnrelu_cat_k(
    const float* __restrict__ t, float* __restrict__ cat, int C, int S,
    int Ctot, int coff, int rep) {
  int c = blockIdx.x, tid = threadIdx.x;
  int N = BATCH * S;
  float s = 0.f, s2 = 0.f;
  for (int i = tid; i < N; i += 256) {
    int b = i / S, sp = i % S;
    float v = t[((size_t)b * C + c) * S + sp];
    s += v; s2 += v * v;
  }
  __shared__ float red[256], red2[256];
  red[tid] = s; red2[tid] = s2;
  __syncthreads();
  for (int st = 128; st > 0; st >>= 1) {
    if (tid < st) { red[tid] += red[tid + st]; red2[tid] += red2[tid + st]; }
    __syncthreads();
  }
  float m = red[0] / N;
  float var = red2[0] / N - m * m;
  float inv = rsqrtf(var + 1e-5f);
  for (int i = tid; i < N; i += 256) {
    int b = i / S, sp = i % S;
    float v = (t[((size_t)b * C + c) * S + sp] - m) * inv;
    v = v > 0.f ? v : 0.f;
    for (int r = 0; r < rep; ++r)
      cat[((size_t)b * Ctot + coff + c * rep + r) * S + sp] = v;
  }
}

// ---------------------------------------------------------------------------
// Final MPS site GEMMs: Mf[s,b,n] = phi_s[b,768] x coresF[s][768,1024] (WMMA).
// Grid: 16 sites x 2 batch-halves x 4 n-quarters.
// ---------------------------------------------------------------------------
__global__ __launch_bounds__(256) void fgemm_k(const float* __restrict__ xinF,
                                               const float* __restrict__ coresF,
                                               float* __restrict__ Mf) {
  __shared__ _Float16 sPhi[32 * 768];  // 48 KB
  int s  = blockIdx.x >> 3;
  int mh = (blockIdx.x >> 2) & 1;
  int nq = blockIdx.x & 3;
  int tid = threadIdx.x, lane = tid & 31, wave = tid >> 5;

  for (int i = tid; i < 32 * 768; i += 256) {
    int br = i / 768, f = i % 768;
    int b = mh * 32 + br;
    int c = f % 256;
    float xv = xinF[((size_t)b * 256 + c) * 16 + s];
    float val = (f < 256) ? xv : ((f < 512) ? (1.0f - xv) : xv * xv);
    sPhi[br * 768 + f] = (_Float16)val;
  }
  __syncthreads();

  const float* Bs = coresF + (size_t)s * 768 * 1024;
  for (int t = wave; t < 32; t += 8) {   // 2 m-tiles x 16 n-tiles
    int mt = t >> 4;
    int nt = t & 15;
    int nb = nq * 256 + nt * 16;
    v8f c = {};
#pragma unroll
    for (int kb = 0; kb < 768; kb += 32) {
      v16h af = lds_a_frag(sPhi, 768, mt * 16, kb, lane);
      v16h bf = glb_b_frag(Bs, 1024, kb, nb, 768, lane);
      c = __builtin_amdgcn_wmma_f32_16x16x32_f16(false, af, false, bf, (short)0,
                                                 c, false, false);
    }
    int n  = nb + (lane & 15);
    int hb = (lane & 16) ? 8 : 0;
#pragma unroll
    for (int v = 0; v < 8; ++v) {
      int bg = mh * 32 + mt * 16 + v + hb;
      Mf[((size_t)s * BATCH + bg) * 1024 + n] = c[v];
    }
  }
}

// ---------------------------------------------------------------------------
// Final chain: r = vlF^T M0 ; r = r M_s (s=1..15) ; out[b,:] = r wrF
// ---------------------------------------------------------------------------
__global__ __launch_bounds__(256) void fchain_k(const float* __restrict__ Mf,
                                                const float* __restrict__ vlF,
                                                const float* __restrict__ wrF,
                                                float* __restrict__ out) {
  int b = blockIdx.x, tid = threadIdx.x;
  __shared__ float r[DBOND], rn[DBOND];
  if (tid < DBOND) {
    float acc = 0.f;
    for (int i = 0; i < DBOND; ++i)
      acc += vlF[i] * Mf[(size_t)b * 1024 + i * DBOND + tid];
    r[tid] = acc;
  }
  __syncthreads();
  for (int s = 1; s < 16; ++s) {
    if (tid < DBOND) {
      float acc = 0.f;
      const float* Ms = Mf + ((size_t)s * BATCH + b) * 1024;
      for (int i = 0; i < DBOND; ++i) acc += r[i] * Ms[i * DBOND + tid];
      rn[tid] = acc;
    }
    __syncthreads();
    if (tid < DBOND) r[tid] = rn[tid];
    __syncthreads();
  }
  for (int o = tid; o < 256; o += 256) {
    float acc = 0.f;
    for (int j = 0; j < DBOND; ++j) acc += r[j] * wrF[j * 256 + o];
    out[b * 256 + o] = acc;
  }
}

// ---------------------------------------------------------------------------
extern "C" void kernel_launch(void* const* d_in, const int* in_sizes, int n_in,
                              void* d_out, int out_size, void* d_ws,
                              size_t ws_size, hipStream_t stream) {
  const float* x      = (const float*)d_in[0];
  const float* cores1 = (const float*)d_in[1];
  const float* vl1    = (const float*)d_in[2];
  const float* wr1    = (const float*)d_in[3];
  const float* cores2 = (const float*)d_in[4];
  const float* vl2    = (const float*)d_in[5];
  const float* wr2    = (const float*)d_in[6];
  const float* cores3 = (const float*)d_in[7];
  const float* vl3    = (const float*)d_in[8];
  const float* wr3    = (const float*)d_in[9];
  const float* coresF = (const float*)d_in[10];
  const float* vlF    = (const float*)d_in[11];
  const float* wrF    = (const float*)d_in[12];
  float* out = (float*)d_out;

  // workspace layout (floats)
  float* W = (float*)d_ws;
  size_t off = 0;
  float* xin1 = W + off; off += (size_t)BATCH * 16 * 4 * 256;   // 1 M
  float* cat1 = W + off; off += (size_t)BATCH * 64 * 256;       // 1 M
  float* xin2 = W + off; off += (size_t)BATCH * 64 * 4 * 64;    // 1 M
  float* cat2 = W + off; off += (size_t)BATCH * 128 * 64;       // 512 K
  float* xin3 = W + off; off += (size_t)BATCH * 128 * 4 * 16;   // 512 K
  float* cat3 = W + off; off += (size_t)BATCH * 256 * 16;       // 256 K
  float* y1   = W + off; off += (size_t)BATCH * 32 * 256;       // 512 K
  float* y2   = W + off; off += (size_t)BATCH * 64 * 64;        // 256 K
  float* y3   = W + off; off += (size_t)BATCH * 128 * 16;       // 128 K
  float* t1   = W + off; off += (size_t)BATCH * 1 * 16 * 16;    // 16 K
  float* t2   = W + off; off += (size_t)BATCH * 64 * 8 * 8;     // 256 K
  float* t3   = W + off; off += (size_t)BATCH * 128 * 4 * 4;    // 128 K
  float* Mf   = W + off; off += (size_t)16 * BATCH * 1024;      // 1 M

  // ---- level 1 ----
  gather1_k<<<(BATCH * 16 * 4 * 256 + 255) / 256, 256, 0, stream>>>(x, xin1);
  mps_level_k<16, 256, 32, 64><<<256, 512, 0, stream>>>(xin1, cores1, vl1, wr1, y1);
  pool_k<<<(BATCH * 1 * 16 * 16 + 255) / 256, 256, 0, stream>>>(x, t1, 1, 128, 128, 8);
  bnrelu_cat_k<<<1, 256, 0, stream>>>(t1, cat1, 1, 256, 64, 0, 32);    // z1 -> ch 0..31
  bn_mps_to_cat_k<<<32, 256, 0, stream>>>(y1, cat1, 32, 256, 64, 32);  // y1 -> ch 32..63

  // ---- level 2 ----
  gather2_k<<<(BATCH * 64 * 4 * 64 + 255) / 256, 256, 0, stream>>>(cat1, xin2);
  mps_level_k<64, 64, 64, 192><<<64, 512, 0, stream>>>(xin2, cores2, vl2, wr2, y2);
  pool_k<<<(BATCH * 64 * 8 * 8 + 255) / 256, 256, 0, stream>>>(cat1, t2, 64, 16, 16, 2);
  bnrelu_cat_k<<<64, 256, 0, stream>>>(t2, cat2, 64, 64, 128, 0, 1);   // z2 -> ch 0..63
  bn_mps_to_cat_k<<<64, 256, 0, stream>>>(y2, cat2, 64, 64, 128, 64);  // y2 -> ch 64..127

  // ---- level 3 ----
  gather3_k<<<(BATCH * 128 * 4 * 16 + 255) / 256, 256, 0, stream>>>(cat2, xin3);
  mps_level_k<128, 16, 128, 384><<<16, 512, 0, stream>>>(xin3, cores3, vl3, wr3, y3);
  pool_k<<<(BATCH * 128 * 4 * 4 + 255) / 256, 256, 0, stream>>>(cat2, t3, 128, 8, 8, 2);
  bnrelu_cat_k<<<128, 256, 0, stream>>>(t3, cat3, 128, 16, 256, 0, 1);   // z3 -> ch 0..127
  bn_mps_to_cat_k<<<128, 256, 0, stream>>>(y3, cat3, 128, 16, 256, 128); // y3 -> ch 128..255

  // ---- final MPS (cat3 reshape (b,256,4,4)->(b,256,16) is identity) ----
  fgemm_k<<<128, 256, 0, stream>>>(cat3, coresF, Mf);
  fchain_k<<<64, 256, 0, stream>>>(Mf, vlF, wrF, out);
}
